// BiDAFAttention_773094113475
// MI455X (gfx1250) — matrix operand order repaired and checked by reading, not compiled, not run on hardware
//
#include <hip/hip_runtime.h>

// ---------------------------------------------------------------------------
// BiDAF attention for MI455X (gfx1250, wave32, WMMA).
// GEMMs run in BF16 WMMA (V_WMMA_F32_16X16X32_BF16) with f32 accumulation.
// ---------------------------------------------------------------------------

typedef __attribute__((ext_vector_type(16))) __bf16 bf16x16;
typedef __attribute__((ext_vector_type(8)))  __bf16 bf16x8;
typedef __attribute__((ext_vector_type(8)))  float  f32x8;

constexpr int B  = 32;
constexpr int LC = 512;
constexpr int LQ = 64;
constexpr int H  = 512;
constexpr int OUTW = 4 * H;           // 2048 output columns
#define NEG_INF_F (-1e30f)

#define DEVI __device__ __forceinline__

DEVI __bf16 to_bf16(float f) { return (__bf16)f; }

// ---- WMMA fragment loaders (CDNA5 ISA 7.12.2 layouts, wave32) -------------
// A (16x32 bf16, row-major, K contiguous): lane r=lane&15 holds row r.
//   lanes 0-15 : VGPR0-3 = K 0..7,  VGPR4-7 = K 16..23
//   lanes 16-31: VGPR0-3 = K 8..15, VGPR4-7 = K 24..31
DEVI bf16x16 load_a_frag(const __bf16* A, int lda, int k, int lane) {
  int r  = lane & 15;
  int hk = (lane >> 4) * 8;
  const __bf16* p = A + (size_t)r * lda + k + hk;
  bf16x8 lo = *(const bf16x8*)(p);        // K = k+hk   .. +7
  bf16x8 hi = *(const bf16x8*)(p + 16);   // K = k+16+hk.. +7
  return __builtin_shufflevector(lo, hi, 0,1,2,3,4,5,6,7,8,9,10,11,12,13,14,15);
}

// B (32x16 bf16). We store B transposed (BT: N x K row-major) so each lane's
// column is contiguous in K: lane n=lane&15 holds column n,
// lanes 0-15 K=k..k+7,k+16..k+23; lanes 16-31 K=k+8..k+15,k+24..k+31 (same as A).
DEVI bf16x16 load_b_frag(const __bf16* BT, int ldb, int k, int lane) {
  int n  = lane & 15;
  int hk = (lane >> 4) * 8;
  const __bf16* p = BT + (size_t)n * ldb + k + hk;
  bf16x8 lo = *(const bf16x8*)(p);
  bf16x8 hi = *(const bf16x8*)(p + 16);
  return __builtin_shufflevector(lo, hi, 0,1,2,3,4,5,6,7,8,9,10,11,12,13,14,15);
}

#define WMMA_BF16(a, b, c) \
  __builtin_amdgcn_wmma_f32_16x16x32_bf16(false, (a), false, (b), (short)0, (c), false, false)

// One wave computes a 32x64 output tile (2x4 WMMA tiles, 64 accum VGPRs).
DEVI void gemm_tile_32x64(const __bf16* __restrict__ A,  int lda,
                          const __bf16* __restrict__ BT, int ldb,
                          int K, int lane, f32x8* acc)
{
  for (int k = 0; k < K; k += 32) {
    if (k + 32 < K) {                    // -> global_prefetch_b8
      __builtin_prefetch(A  + (size_t)(lane & 15) * lda + k + 32, 0, 1);
      __builtin_prefetch(BT + (size_t)(lane & 15) * ldb + k + 32, 0, 1);
    }
    bf16x16 a0 = load_a_frag(A,                      lda, k, lane);
    bf16x16 a1 = load_a_frag(A + (size_t)16 * lda,   lda, k, lane);
    bf16x16 b0 = load_b_frag(BT,                     ldb, k, lane);
    bf16x16 b1 = load_b_frag(BT + (size_t)16 * ldb,  ldb, k, lane);
    bf16x16 b2 = load_b_frag(BT + (size_t)32 * ldb,  ldb, k, lane);
    bf16x16 b3 = load_b_frag(BT + (size_t)48 * ldb,  ldb, k, lane);
    acc[0] = WMMA_BF16(a0, b0, acc[0]);
    acc[1] = WMMA_BF16(a0, b1, acc[1]);
    acc[2] = WMMA_BF16(a0, b2, acc[2]);
    acc[3] = WMMA_BF16(a0, b3, acc[3]);
    acc[4] = WMMA_BF16(a1, b0, acc[4]);
    acc[5] = WMMA_BF16(a1, b1, acc[5]);
    acc[6] = WMMA_BF16(a1, b2, acc[6]);
    acc[7] = WMMA_BF16(a1, b3, acc[7]);
  }
}

// ---------------------------------------------------------------------------
// Prep kernels
// ---------------------------------------------------------------------------

// One wave per (b,i): s0 = dot(c_row, c_weight); cw_bf = bf16(c_row * cq_w).
__global__ void prep_c_kernel(const float* __restrict__ c,
                              const float* __restrict__ c_weight,
                              const float* __restrict__ cq_weight,
                              __bf16* __restrict__ cw_bf,
                              float* __restrict__ s0)
{
  int wid  = (blockIdx.x * blockDim.x + threadIdx.x) >> 5;
  int lane = threadIdx.x & 31;
  if (wid >= B * LC) return;
  const float* row = c + (size_t)wid * H;
  __bf16* orow = cw_bf + (size_t)wid * H;
  float acc = 0.f;
  for (int h = lane; h < H; h += 32) {
    float cv = row[h];
    acc += cv * c_weight[h];
    orow[h] = to_bf16(cv * cq_weight[h]);
  }
  #pragma unroll
  for (int off = 16; off; off >>= 1) acc += __shfl_xor(acc, off, 32);
  if (lane == 0) s0[wid] = acc;
}

// One wave per (b,j): s1 = dot(q_row, q_weight); q_bf = bf16(q_row).
__global__ void prep_q_kernel(const float* __restrict__ q,
                              const float* __restrict__ q_weight,
                              __bf16* __restrict__ q_bf,
                              float* __restrict__ s1)
{
  int wid  = (blockIdx.x * blockDim.x + threadIdx.x) >> 5;
  int lane = threadIdx.x & 31;
  if (wid >= B * LQ) return;
  const float* row = q + (size_t)wid * H;
  __bf16* orow = q_bf + (size_t)wid * H;
  float acc = 0.f;
  for (int h = lane; h < H; h += 32) {
    float qv = row[h];
    acc += qv * q_weight[h];
    orow[h] = to_bf16(qv);
  }
  #pragma unroll
  for (int off = 16; off; off >>= 1) acc += __shfl_xor(acc, off, 32);
  if (lane == 0) s1[wid] = acc;
}

// LDS-tiled transpose + f32->bf16: src (b, R, C) f32 -> dst (b, C, R) bf16.
// block (32,8), grid (C/32, R/32, B)
__global__ void transpose_to_bf16(const float* __restrict__ src,
                                  __bf16* __restrict__ dst, int R, int C)
{
  __shared__ float tile[32][33];
  int b  = blockIdx.z;
  int c0 = blockIdx.x * 32;
  int r0 = blockIdx.y * 32;
  const float* s = src + (size_t)b * R * C;
  __bf16*      d = dst + (size_t)b * R * C;
  #pragma unroll
  for (int t = threadIdx.y; t < 32; t += 8)
    tile[t][threadIdx.x] = s[(size_t)(r0 + t) * C + c0 + threadIdx.x];
  __syncthreads();
  #pragma unroll
  for (int t = threadIdx.y; t < 32; t += 8)
    d[(size_t)(c0 + t) * R + r0 + threadIdx.x] = to_bf16(tile[threadIdx.x][t]);
}

// ---------------------------------------------------------------------------
// s = cw @ q^T + s0[i] + s1[j] + bias        (M=LC, N=LQ, K=H)
// ---------------------------------------------------------------------------
__global__ __launch_bounds__(128)
void gemm_s_kernel(const __bf16* __restrict__ cw_bf,
                   const __bf16* __restrict__ q_bf,
                   const float* __restrict__ s0,
                   const float* __restrict__ s1,
                   const float* __restrict__ bias,
                   float* __restrict__ s)
{
  int lane = threadIdx.x & 31;
  int wave = threadIdx.x >> 5;
  int b  = blockIdx.z;
  int m0 = (blockIdx.y * 4 + wave) * 32;
  int n0 = blockIdx.x * 64;
  const __bf16* A  = cw_bf + (size_t)b * LC * H + (size_t)m0 * H;
  const __bf16* BT = q_bf  + (size_t)b * LQ * H + (size_t)n0 * H;
  f32x8 acc[8] = {};
  gemm_tile_32x64(A, H, BT, H, H, lane, acc);

  float bv = bias[0];
  int n  = lane & 15;
  int mb = (lane >> 4) * 8;
  #pragma unroll
  for (int mi = 0; mi < 2; ++mi)
    #pragma unroll
    for (int ni = 0; ni < 4; ++ni) {
      int j = n0 + ni * 16 + n;
      float s1v = s1[b * LQ + j] + bv;
      #pragma unroll
      for (int v = 0; v < 8; ++v) {
        int i = m0 + mi * 16 + mb + v;
        s[((size_t)b * LC + i) * LQ + j] = acc[mi * 4 + ni][v] + s0[b * LC + i] + s1v;
      }
    }
}

// ---------------------------------------------------------------------------
// Masked softmaxes
// ---------------------------------------------------------------------------

// axis=2 (rows of length LQ=64): one wave per (b,i), 2 elems/lane.
__global__ void softmax_rows_kernel(const float* __restrict__ s,
                                    const unsigned char* __restrict__ q_mask,
                                    __bf16* __restrict__ a1)
{
  int wid  = (blockIdx.x * blockDim.x + threadIdx.x) >> 5;
  int lane = threadIdx.x & 31;
  if (wid >= B * LC) return;
  int b = wid / LC;
  const float* row = s + (size_t)wid * LQ;
  const unsigned char* qm = q_mask + (size_t)b * LQ;
  float l0 = qm[lane]      ? row[lane]      : NEG_INF_F;
  float l1 = qm[lane + 32] ? row[lane + 32] : NEG_INF_F;
  float m = fmaxf(l0, l1);
  #pragma unroll
  for (int off = 16; off; off >>= 1) m = fmaxf(m, __shfl_xor(m, off, 32));
  float e0 = __expf(l0 - m), e1 = __expf(l1 - m);
  float sum = e0 + e1;
  #pragma unroll
  for (int off = 16; off; off >>= 1) sum += __shfl_xor(sum, off, 32);
  float inv = 1.f / sum;
  __bf16* orow = a1 + (size_t)wid * LQ;
  orow[lane]      = to_bf16(e0 * inv);
  orow[lane + 32] = to_bf16(e1 * inv);
}

// axis=1 (columns of length LC=512): one thread per (b,j), coalesced rows.
__global__ void softmax_cols_kernel(const float* __restrict__ s,
                                    const unsigned char* __restrict__ c_mask,
                                    __bf16* __restrict__ a2)
{
  int b = blockIdx.x;
  int j = threadIdx.x;           // 0..63
  const float* sb = s + (size_t)b * LC * LQ;
  const unsigned char* cm = c_mask + (size_t)b * LC;
  float m = NEG_INF_F;
  for (int i = 0; i < LC; ++i) {
    float l = cm[i] ? sb[(size_t)i * LQ + j] : NEG_INF_F;
    m = fmaxf(m, l);
  }
  float sum = 0.f;
  for (int i = 0; i < LC; ++i) {
    float l = cm[i] ? sb[(size_t)i * LQ + j] : NEG_INF_F;
    sum += __expf(l - m);
  }
  float inv = 1.f / sum;
  __bf16* ab = a2 + (size_t)b * LC * LQ;
  for (int i = 0; i < LC; ++i) {
    float l = cm[i] ? sb[(size_t)i * LQ + j] : NEG_INF_F;
    ab[(size_t)i * LQ + j] = to_bf16(__expf(l - m) * inv);
  }
}

// ---------------------------------------------------------------------------
// a = a1 @ q  (M=LC, N=H, K=LQ) -> written straight into out[:, :, H:2H]
// ---------------------------------------------------------------------------
__global__ __launch_bounds__(128)
void gemm_a_kernel(const __bf16* __restrict__ a1_bf,
                   const __bf16* __restrict__ qT_bf,
                   float* __restrict__ out)
{
  int lane = threadIdx.x & 31;
  int wave = threadIdx.x >> 5;
  int b  = blockIdx.z;
  int m0 = (blockIdx.y * 4 + wave) * 32;
  int n0 = blockIdx.x * 64;
  const __bf16* A  = a1_bf + (size_t)b * LC * LQ + (size_t)m0 * LQ;
  const __bf16* BT = qT_bf + (size_t)b * H  * LQ + (size_t)n0 * LQ;
  f32x8 acc[8] = {};
  gemm_tile_32x64(A, LQ, BT, LQ, LQ, lane, acc);

  int n  = lane & 15;
  int mb = (lane >> 4) * 8;
  #pragma unroll
  for (int mi = 0; mi < 2; ++mi)
    #pragma unroll
    for (int ni = 0; ni < 4; ++ni) {
      int j = n0 + ni * 16 + n;
      #pragma unroll
      for (int v = 0; v < 8; ++v) {
        int i = m0 + mi * 16 + mb + v;
        out[((size_t)b * LC + i) * OUTW + H + j] = acc[mi * 4 + ni][v];
      }
    }
}

// ---------------------------------------------------------------------------
// bmat = a1 @ a2^T  (M=LC, N=LC, K=LQ), stored bf16 for the next GEMM
// ---------------------------------------------------------------------------
__global__ __launch_bounds__(128)
void gemm_bmat_kernel(const __bf16* __restrict__ a1_bf,
                      const __bf16* __restrict__ a2_bf,
                      __bf16* __restrict__ bmat_bf)
{
  int lane = threadIdx.x & 31;
  int wave = threadIdx.x >> 5;
  int b  = blockIdx.z;
  int m0 = (blockIdx.y * 4 + wave) * 32;
  int n0 = blockIdx.x * 64;
  const __bf16* A  = a1_bf + (size_t)b * LC * LQ + (size_t)m0 * LQ;
  const __bf16* BT = a2_bf + (size_t)b * LC * LQ + (size_t)n0 * LQ;
  f32x8 acc[8] = {};
  gemm_tile_32x64(A, LQ, BT, LQ, LQ, lane, acc);

  int n  = lane & 15;
  int mb = (lane >> 4) * 8;
  #pragma unroll
  for (int mi = 0; mi < 2; ++mi)
    #pragma unroll
    for (int ni = 0; ni < 4; ++ni) {
      int j = n0 + ni * 16 + n;
      #pragma unroll
      for (int v = 0; v < 8; ++v) {
        int i = m0 + mi * 16 + mb + v;
        bmat_bf[((size_t)b * LC + i) * LC + j] = to_bf16(acc[mi * 4 + ni][v]);
      }
    }
}

// ---------------------------------------------------------------------------
// bres = bmat @ c  (M=LC, N=H, K=LC) — dominant GEMM (8.6 GFLOP total)
// ---------------------------------------------------------------------------
__global__ __launch_bounds__(128)
void gemm_b_kernel(const __bf16* __restrict__ bmat_bf,
                   const __bf16* __restrict__ cT_bf,
                   float* __restrict__ bres)
{
  int lane = threadIdx.x & 31;
  int wave = threadIdx.x >> 5;
  int b  = blockIdx.z;
  int m0 = (blockIdx.y * 4 + wave) * 32;
  int n0 = blockIdx.x * 64;
  const __bf16* A  = bmat_bf + (size_t)b * LC * LC + (size_t)m0 * LC;
  const __bf16* BT = cT_bf   + (size_t)b * H  * LC + (size_t)n0 * LC;
  f32x8 acc[8] = {};
  gemm_tile_32x64(A, LC, BT, LC, LC, lane, acc);

  int n  = lane & 15;
  int mb = (lane >> 4) * 8;
  #pragma unroll
  for (int mi = 0; mi < 2; ++mi)
    #pragma unroll
    for (int ni = 0; ni < 4; ++ni) {
      int j = n0 + ni * 16 + n;
      #pragma unroll
      for (int v = 0; v < 8; ++v) {
        int i = m0 + mi * 16 + mb + v;
        bres[((size_t)b * LC + i) * H + j] = acc[mi * 4 + ni][v];
      }
    }
}

// ---------------------------------------------------------------------------
// out = [c, a, c*a, c*b] ; 'a' slice already written by gemm_a_kernel.
// ---------------------------------------------------------------------------
__global__ void finalize_kernel(const float* __restrict__ c,
                                const float* __restrict__ bres,
                                float* __restrict__ out)
{
  size_t idx = (size_t)blockIdx.x * blockDim.x + threadIdx.x;
  if (idx >= (size_t)B * LC * H) return;
  size_t row = idx / H;
  int h = (int)(idx % H);
  float cv = c[idx];
  float bv = bres[idx];
  float* orow = out + row * OUTW;
  float av = orow[H + h];
  orow[h]         = cv;
  orow[2 * H + h] = cv * av;
  orow[3 * H + h] = cv * bv;
}

// ---------------------------------------------------------------------------
// Launch
// ---------------------------------------------------------------------------
extern "C" void kernel_launch(void* const* d_in, const int* in_sizes, int n_in,
                              void* d_out, int out_size, void* d_ws, size_t ws_size,
                              hipStream_t stream) {
  const float* c         = (const float*)d_in[0];
  const float* q         = (const float*)d_in[1];
  const unsigned char* c_mask = (const unsigned char*)d_in[2];
  const unsigned char* q_mask = (const unsigned char*)d_in[3];
  const float* c_weight  = (const float*)d_in[4];
  const float* q_weight  = (const float*)d_in[5];
  const float* cq_weight = (const float*)d_in[6];
  const float* bias      = (const float*)d_in[7];
  float* out = (float*)d_out;

  // Workspace layout (bytes)
  char* ws = (char*)d_ws;
  size_t off = 0;
  __bf16* cw_bf   = (__bf16*)(ws + off); off += (size_t)B * LC * H * 2;   // 16.8 MB
  __bf16* cT_bf   = (__bf16*)(ws + off); off += (size_t)B * LC * H * 2;   // 16.8 MB
  __bf16* q_bf    = (__bf16*)(ws + off); off += (size_t)B * LQ * H * 2;   //  2.1 MB
  __bf16* qT_bf   = (__bf16*)(ws + off); off += (size_t)B * LQ * H * 2;   //  2.1 MB
  float*  s0      = (float*) (ws + off); off += (size_t)B * LC * 4;
  float*  s1      = (float*) (ws + off); off += (size_t)B * LQ * 4;
  float*  s       = (float*) (ws + off); off += (size_t)B * LC * LQ * 4;  //  4.2 MB
  __bf16* a1_bf   = (__bf16*)(ws + off); off += (size_t)B * LC * LQ * 2;
  __bf16* a2_bf   = (__bf16*)(ws + off); off += (size_t)B * LC * LQ * 2;
  __bf16* bmat_bf = (__bf16*)(ws + off); off += (size_t)B * LC * LC * 2;  // 16.8 MB
  float*  bres    = (float*) (ws + off); off += (size_t)B * LC * H * 4;   // 33.6 MB

  // 1) GEMVs + bf16 packing
  prep_c_kernel<<<(B * LC) / 8, 256, 0, stream>>>(c, c_weight, cq_weight, cw_bf, s0);
  prep_q_kernel<<<(B * LQ) / 8, 256, 0, stream>>>(q, q_weight, q_bf, s1);

  // 2) bf16 transposes (K-contiguous B operands for WMMA)
  transpose_to_bf16<<<dim3(H / 32, LC / 32, B), dim3(32, 8), 0, stream>>>(c, cT_bf, LC, H);
  transpose_to_bf16<<<dim3(H / 32, LQ / 32, B), dim3(32, 8), 0, stream>>>(q, qT_bf, LQ, H);

  // 3) s = cw @ q^T + biases   (WMMA)
  gemm_s_kernel<<<dim3(LQ / 64, LC / 128, B), 128, 0, stream>>>(cw_bf, q_bf, s0, s1, bias, s);

  // 4) masked softmaxes -> bf16
  softmax_rows_kernel<<<(B * LC) / 8, 256, 0, stream>>>(s, q_mask, a1_bf);
  softmax_cols_kernel<<<B, LQ, 0, stream>>>(s, c_mask, a2_bf);

  // 5) a = a1 @ q  (into out slice)   (WMMA)
  gemm_a_kernel<<<dim3(H / 64, LC / 128, B), 128, 0, stream>>>(a1_bf, qT_bf, out);

  // 6) bmat = a1 @ a2^T   (WMMA)
  gemm_bmat_kernel<<<dim3(LC / 64, LC / 128, B), 128, 0, stream>>>(a1_bf, a2_bf, bmat_bf);

  // 7) bres = bmat @ c   (WMMA, dominant)
  gemm_b_kernel<<<dim3(H / 64, LC / 128, B), 128, 0, stream>>>(bmat_bf, cT_bf, bres);

  // 8) out = [c, a, c*a, c*b]
  finalize_kernel<<<(int)(((size_t)B * LC * H + 255) / 256), 256, 0, stream>>>(c, bres, out);
}
